// TFMBSLinear_2465311228218
// MI455X (gfx1250) — 8-bit matrix operand order repaired and checked by reading, compiled, not run on hardware
//
#include <hip/hip_runtime.h>


typedef __attribute__((ext_vector_type(8))) int   v8i;
typedef __attribute__((ext_vector_type(4))) float v4f;
typedef __attribute__((ext_vector_type(2))) int   v2i;

#define IN_F   8192
#define OUT_F  8192
#define BATCH  8
#define WAVES_PER_BLOCK 8
#define BLOCK_THREADS   (WAVES_PER_BLOCK * 32)
#define KCHUNK (IN_F / WAVES_PER_BLOCK)

__global__ __launch_bounds__(512) void quantize_x_kernel(
    const float* __restrict__ x, signed char* __restrict__ sx) {
  int idx = blockIdx.x * blockDim.x + threadIdx.x;
  int row = idx >> 13;
  signed char s = 0;
  if (row < BATCH) {
    float v = x[idx];
    s = (signed char)((v > 0.0f) - (v < 0.0f));
  }
  sx[idx] = s;
}

__device__ __forceinline__ int sgnb(float f) {
  int s = __float_as_int(f);
  int t = (s >> 31) | 1;
  return (f != 0.0f) ? t : 0;
}

__device__ __forceinline__ int sgn_pack4(v4f f) {
  int b0 = sgnb(f.x), b1 = sgnb(f.y), b2 = sgnb(f.z), b3 = sgnb(f.w);
  return (b0 & 0xff) | ((b1 & 0xff) << 8) | ((b2 & 0xff) << 16) | (b3 << 24);
}

__device__ __forceinline__ v8i k64_step(const signed char* aptr,
                                        const float* wptr, v8i acc) {
  const v2i* ap = (const v2i*)aptr;
  v2i a01 = ap[0];
  v2i a23 = ap[1];
  v2i a45 = ap[4];
  v2i a67 = ap[5];
  v8i A;
  A[0] = a01.x; A[1] = a01.y; A[2] = a23.x; A[3] = a23.y;
  A[4] = a45.x; A[5] = a45.y; A[6] = a67.x; A[7] = a67.y;

  const v4f* wp = (const v4f*)wptr;
  v8i Bm;
  #pragma unroll
  for (int j = 0; j < 4; ++j) {
    v4f f = __builtin_nontemporal_load(&wp[j]);
    Bm[j] = sgn_pack4(f);
  }
  #pragma unroll
  for (int j = 0; j < 4; ++j) {
    v4f f = __builtin_nontemporal_load(&wp[8 + j]);
    Bm[4 + j] = sgn_pack4(f);
  }

  return __builtin_amdgcn_wmma_i32_16x16x64_iu8(
       true, A,  true, Bm, acc,
       false,  false);
}

__global__ __launch_bounds__(BLOCK_THREADS) void ternary_gemv_wmma(
    const float* __restrict__ W,
    const float* __restrict__ bias,
    const signed char* __restrict__ sx,
    float* __restrict__ out)
{
  const int lane   = threadIdx.x & 31;
  const int wave   = threadIdx.x >> 5;
  const int o_base = blockIdx.x * 16;

  const int m  = lane & 15;
  const int hi = lane >> 4;

  const signed char* aptr = sx + m * IN_F + wave * KCHUNK + hi * 16;
  const float* wptr = W + (size_t)(o_base + m) * IN_F + wave * KCHUNK + hi * 16;

  v8i acc0 = (v8i)0;
  v8i acc1 = (v8i)0;

  for (int it = 0; it < KCHUNK / 128; ++it) {
    acc0 = k64_step(aptr,      wptr,      acc0);
    acc1 = k64_step(aptr + 64, wptr + 64, acc1);
    aptr += 128;
    wptr += 128;
  }
  v8i acc = acc0 + acc1;

  __shared__ int red[WAVES_PER_BLOCK][8][32];
  #pragma unroll
  for (int r = 0; r < 8; ++r) red[wave][r][lane] = acc[r];
  __syncthreads();

  if (wave == 0 && lane < 16) {
    const float bv = bias[o_base + lane];
    #pragma unroll
    for (int r = 0; r < 8; ++r) {
      int s = 0;
      #pragma unroll
      for (int w = 0; w < WAVES_PER_BLOCK; ++w) s += red[w][r][lane];
      out[r * OUT_F + o_base + lane] = (float)s + bv;
    }
  }
}

extern "C" void kernel_launch(void* const* d_in, const int* in_sizes, int n_in,
                              void* d_out, int out_size, void* d_ws, size_t ws_size,
                              hipStream_t stream) {
  const float* x    = (const float*)d_in[0];
  const float* W    = (const float*)d_in[1];
  const float* bias = (const float*)d_in[2];
  float* out        = (float*)d_out;

  signed char* sx = (signed char*)d_ws;

  quantize_x_kernel<<<(16 * IN_F) / 512, 512, 0, stream>>>(x, sx);
  ternary_gemv_wmma<<<OUT_F / 16, BLOCK_THREADS, 0, stream>>>(W, bias, sx, out);
}
